// VGAEModel_68874095558957
// MI455X (gfx1250) — matrix operand order repaired and checked by reading, not compiled, not run on hardware
//
#include <hip/hip_runtime.h>

typedef __attribute__((ext_vector_type(16))) _Float16 v16h;
typedef __attribute__((ext_vector_type(8)))  _Float16 h8;
typedef __attribute__((ext_vector_type(8)))  float    v8f;

#define NN   8192
#define EE   131072
#define DIN  512
#define HID1 256
#define HID2 64

// ---------------------------------------------------------------- utility

__global__ void k_init(float* __restrict__ deg, double* __restrict__ acc, int n) {
  int i = blockIdx.x * blockDim.x + threadIdx.x;
  if (i == 0) *acc = 0.0;                 // BCE accumulator
  if (i < n) deg[i] = 1.0f;               // self-loop weight
}

__global__ void k_deg_edges(const int* __restrict__ dst, const float* __restrict__ w,
                            float* __restrict__ deg, int e) {
  int i = blockIdx.x * blockDim.x + threadIdx.x;
  if (i < e) atomicAdd(&deg[dst[i]], w[i]);
}

__global__ void k_dinv(const float* __restrict__ deg, float* __restrict__ dinv, int n) {
  int i = blockIdx.x * blockDim.x + threadIdx.x;
  if (i < n) dinv[i] = rsqrtf(fmaxf(deg[i], 1e-12f));   // deg >= 1 always
}

__global__ void k_cvt_f16(const float* __restrict__ in, _Float16* __restrict__ out, int n) {
  int i = blockIdx.x * blockDim.x + threadIdx.x;
  if (i < n) out[i] = (_Float16)in[i];
}

// W [K x Nc] f32 row-major -> Wt [Nc x K] f16 (so GEMM B-fragment loads are contiguous)
__global__ void k_transpose_f16(const float* __restrict__ w, _Float16* __restrict__ wt,
                                int K, int Nc) {
  int idx = blockIdx.x * blockDim.x + threadIdx.x;
  if (idx >= K * Nc) return;
  int k = idx / Nc, n = idx % Nc;
  wt[(size_t)n * K + k] = (_Float16)w[idx];
}

// ---------------------------------------------------------------- WMMA GEMM
// C[M,Nc] = A[M,K] * B[K,Nc];  A f16 row-major, Bt f16 [Nc x K] row-major, C f32.
// One 16x16 output tile per wave, K stepped by 32 (v_wmma_f32_16x16x32_f16).
__global__ void __launch_bounds__(256)
k_wmma_gemm(const _Float16* __restrict__ A, const _Float16* __restrict__ Bt,
            float* __restrict__ C, int M, int Nc, int K) {
  int wave = blockIdx.x * (blockDim.x >> 5) + (threadIdx.x >> 5);
  int lane = threadIdx.x & 31;
  int l16  = lane & 15;
  int h    = lane >> 4;
  int tilesN = Nc >> 4;
  int tm = wave / tilesN;
  int tn = wave % tilesN;
  if (tm * 16 >= M) return;               // wave-uniform

  const _Float16* arow = A  + (size_t)(tm * 16 + l16) * K;   // A: lane holds row M=l16
  const _Float16* brow = Bt + (size_t)(tn * 16 + l16) * K;   // B: lane holds col N=l16
  v8f c = {};
  for (int k0 = 0; k0 < K; k0 += 32) {
    union { v16h v; h8 p[2]; } a, b;
    // A slots 0..7: K = k0+8h .. ; slots 8..15: K = k0+16+8h ..
    a.p[0] = *(const h8*)(arow + k0 + 8 * h);
    a.p[1] = *(const h8*)(arow + k0 + 16 + 8 * h);
    // B slots 0..7: K = k0+8h .. ; slots 8..15: K = k0+16+8h ..
    b.p[0] = *(const h8*)(brow + k0 + 8 * h);
    b.p[1] = *(const h8*)(brow + k0 + 16 + 8 * h);
    c = __builtin_amdgcn_wmma_f32_16x16x32_f16(false, a.v, false, b.v,
                                               (short)0, c, false, false);
  }
#pragma unroll
  for (int r = 0; r < 8; ++r)
    C[(size_t)(tm * 16 + r + 8 * h) * Nc + tn * 16 + l16] = c[r];
}

// ---------------------------------------------------------------- GCN aggregation
// agg[i] = b + dinv[i]^2 * h[i]   (self-loop term + bias)
__global__ void k_agg_init(const float* __restrict__ h, const float* __restrict__ dinv,
                           const float* __restrict__ bias, float* __restrict__ agg, int F) {
  int i = blockIdx.x, f = threadIdx.x;
  float di = dinv[i];
  agg[(size_t)i * F + f] = bias[f] + di * di * h[(size_t)i * F + f];
}

// agg[dst] += dinv[src]*w*dinv[dst] * h[src]   (one block per edge, lane per feature)
__global__ void k_agg_edges(const int* __restrict__ src, const int* __restrict__ dst,
                            const float* __restrict__ w, const float* __restrict__ dinv,
                            const float* __restrict__ h, float* __restrict__ agg, int F) {
  int e = blockIdx.x, f = threadIdx.x;
  int s = src[e], d = dst[e];
  float coef = dinv[s] * w[e] * dinv[d];
  atomicAdd(&agg[(size_t)d * F + f], coef * h[(size_t)s * F + f]);
}

__global__ void k_relu_cvt(float* __restrict__ x, _Float16* __restrict__ xh, int n) {
  int i = blockIdx.x * blockDim.x + threadIdx.x;
  if (i >= n) return;
  float v = fmaxf(x[i], 0.0f);
  x[i] = v;
  xh[i] = (_Float16)v;
}

// z = mean + noise * exp(log_std); write f32 z to output and f16 copy for decoder GEMM
__global__ void k_z(const float* __restrict__ mean, const float* __restrict__ logstd,
                    const float* __restrict__ noise, float* __restrict__ zout,
                    _Float16* __restrict__ zh, int n) {
  int i = blockIdx.x * blockDim.x + threadIdx.x;
  if (i >= n) return;
  float z = mean[i] + noise[i] * expf(logstd[i]);
  zout[i] = z;
  zh[i] = (_Float16)z;
}

// ---------------------------------------------------------------- fused z@z^T + softplus sum
// logits is symmetric: process only tiles with ti <= tj, double off-diagonal tile sums.
// Each live wave: one 16x16 tile (K=64, 2 WMMAs), stable softplus in regs, wave-reduce,
// one f64 atomic. Never materializes the 8192x8192 matrix; halves GEMM + transcendentals.
__global__ void __launch_bounds__(256)
k_bce_tiles(const _Float16* __restrict__ zh, double* __restrict__ acc) {
  int wave = blockIdx.x * 8 + (threadIdx.x >> 5);
  int ti = wave >> 9;          // 512 tiles per dim
  int tj = wave & 511;
  if (ti > tj) return;         // wave-uniform: symmetry — lower triangle skipped
  int lane = threadIdx.x & 31;
  int l16  = lane & 15;
  int h    = lane >> 4;
  const _Float16* arow = zh + (size_t)(ti * 16 + l16) * HID2;
  const _Float16* brow = zh + (size_t)(tj * 16 + l16) * HID2;  // B col j == z row j
  v8f c = {};
#pragma unroll
  for (int k0 = 0; k0 < HID2; k0 += 32) {
    union { v16h v; h8 p[2]; } a, b;
    a.p[0] = *(const h8*)(arow + k0 + 8 * h);
    a.p[1] = *(const h8*)(arow + k0 + 16 + 8 * h);
    b.p[0] = *(const h8*)(brow + k0 + 8 * h);
    b.p[1] = *(const h8*)(brow + k0 + 16 + 8 * h);
    c = __builtin_amdgcn_wmma_f32_16x16x32_f16(false, a.v, false, b.v,
                                               (short)0, c, false, false);
  }
  float s = 0.0f;
#pragma unroll
  for (int r = 0; r < 8; ++r) {
    float l = c[r];
    s += fmaxf(l, 0.0f) + log1pf(expf(-fabsf(l)));   // stable softplus(l)
  }
  if (ti < tj) s *= 2.0f;      // off-diagonal tile counts for (i,j) and (j,i)
#pragma unroll
  for (int off = 16; off > 0; off >>= 1) s += __shfl_xor(s, off, 32);
  if (lane == 0) atomicAdd(acc, (double)s);
}

// Edge correction: softplus(-l) - softplus(l) = -l  =>  subtract l over edges
__global__ void k_bce_edges(const int* __restrict__ src, const int* __restrict__ dst,
                            const float* __restrict__ z, double* __restrict__ acc, int e) {
  int i = blockIdx.x * blockDim.x + threadIdx.x;
  if (i >= e) return;
  int s = src[i], d = dst[i];
  const float* zs = z + (size_t)s * HID2;
  const float* zd = z + (size_t)d * HID2;
  float l = 0.0f;
#pragma unroll
  for (int k = 0; k < HID2; ++k) l += zs[k] * zd[k];
  atomicAdd(acc, (double)(-l));
}

__global__ void k_final(const double* __restrict__ acc, float* __restrict__ out) {
  if (threadIdx.x == 0 && blockIdx.x == 0) out[0] = (float)(*acc);
}

// ---------------------------------------------------------------- launch

extern "C" void kernel_launch(void* const* d_in, const int* in_sizes, int n_in,
                              void* d_out, int out_size, void* d_ws, size_t ws_size,
                              hipStream_t stream) {
  const float* x     = (const float*)d_in[0];
  const int*   ei    = (const int*)  d_in[1];      // [2, E]
  const float* ew    = (const float*)d_in[2];
  const float* noise = (const float*)d_in[3];
  const float* W1    = (const float*)d_in[4];
  const float* b1    = (const float*)d_in[5];
  const float* W2    = (const float*)d_in[6];
  const float* b2    = (const float*)d_in[7];
  const float* W3    = (const float*)d_in[8];
  const float* b3    = (const float*)d_in[9];
  const int* src = ei;
  const int* dst = ei + EE;

  float* out_bce = (float*)d_out;
  float* out_z   = (float*)d_out + 1;              // [N, H2] f32

  // workspace carve-up (256B aligned)
  char* ws = (char*)d_ws;
  size_t off = 0;
  auto carve = [&](size_t bytes) -> char* {
    char* p = ws + off;
    off += (bytes + 255) & ~(size_t)255;
    return p;
  };
  float*    deg    = (float*)   carve((size_t)NN * 4);
  float*    dinv   = (float*)   carve((size_t)NN * 4);
  double*   acc    = (double*)  carve(8);
  _Float16* w1t    = (_Float16*)carve((size_t)HID1 * DIN * 2);
  _Float16* w2t    = (_Float16*)carve((size_t)HID2 * HID1 * 2);
  _Float16* w3t    = (_Float16*)carve((size_t)HID2 * HID1 * 2);
  _Float16* xh     = (_Float16*)carve((size_t)NN * DIN * 2);
  float*    xw     = (float*)   carve((size_t)NN * HID1 * 4);
  float*    hf     = (float*)   carve((size_t)NN * HID1 * 4);
  _Float16* hh     = (_Float16*)carve((size_t)NN * HID1 * 2);
  float*    xw2    = (float*)   carve((size_t)NN * HID2 * 4);
  float*    xw3    = (float*)   carve((size_t)NN * HID2 * 4);
  float*    meanb  = (float*)   carve((size_t)NN * HID2 * 4);
  float*    logstd = (float*)   carve((size_t)NN * HID2 * 4);
  _Float16* zh     = (_Float16*)carve((size_t)NN * HID2 * 2);
  (void)ws_size; (void)n_in; (void)in_sizes; (void)out_size;

  // degrees + normalization
  k_init<<<(NN + 255) / 256, 256, 0, stream>>>(deg, acc, NN);
  k_deg_edges<<<(EE + 255) / 256, 256, 0, stream>>>(dst, ew, deg, EE);
  k_dinv<<<(NN + 255) / 256, 256, 0, stream>>>(deg, dinv, NN);

  // weight prep (f16, transposed) + x -> f16
  k_transpose_f16<<<(DIN * HID1 + 255) / 256, 256, 0, stream>>>(W1, w1t, DIN, HID1);
  k_transpose_f16<<<(HID1 * HID2 + 255) / 256, 256, 0, stream>>>(W2, w2t, HID1, HID2);
  k_transpose_f16<<<(HID1 * HID2 + 255) / 256, 256, 0, stream>>>(W3, w3t, HID1, HID2);
  k_cvt_f16<<<(NN * DIN + 255) / 256, 256, 0, stream>>>(x, xh, NN * DIN);

  // layer 1: xw = x@W1 ; agg ; relu -> hf (f32) + hh (f16)
  k_wmma_gemm<<<(NN / 16) * (HID1 / 16) / 8, 256, 0, stream>>>(xh, w1t, xw, NN, HID1, DIN);
  k_agg_init<<<NN, HID1, 0, stream>>>(xw, dinv, b1, hf, HID1);
  k_agg_edges<<<EE, HID1, 0, stream>>>(src, dst, ew, dinv, xw, hf, HID1);
  k_relu_cvt<<<(NN * HID1 + 255) / 256, 256, 0, stream>>>(hf, hh, NN * HID1);

  // layer 2/3: mean, log_std
  k_wmma_gemm<<<(NN / 16) * (HID2 / 16) / 8, 256, 0, stream>>>(hh, w2t, xw2, NN, HID2, HID1);
  k_wmma_gemm<<<(NN / 16) * (HID2 / 16) / 8, 256, 0, stream>>>(hh, w3t, xw3, NN, HID2, HID1);
  k_agg_init<<<NN, HID2, 0, stream>>>(xw2, dinv, b2, meanb, HID2);
  k_agg_edges<<<EE, HID2, 0, stream>>>(src, dst, ew, dinv, xw2, meanb, HID2);
  k_agg_init<<<NN, HID2, 0, stream>>>(xw3, dinv, b3, logstd, HID2);
  k_agg_edges<<<EE, HID2, 0, stream>>>(src, dst, ew, dinv, xw3, logstd, HID2);

  // reparameterize
  k_z<<<(NN * HID2 + 255) / 256, 256, 0, stream>>>(meanb, logstd, noise, out_z, zh, NN * HID2);

  // fused decoder: 2*sum_{upper} softplus + diag, minus sum of logits over edges
  k_bce_tiles<<<(NN / 16) * (NN / 16) / 8, 256, 0, stream>>>(zh, acc);
  k_bce_edges<<<(EE + 255) / 256, 256, 0, stream>>>(src, dst, out_z, acc, EE);
  k_final<<<1, 1, 0, stream>>>(acc, out_bce);
}
